// VoxelProposalLayer_61735859912944
// MI455X (gfx1250) — compile-verified, matrix operand order checked by reading, not run on hardware
//
#include <hip/hip_runtime.h>
#include <hip/hip_bf16.h>
#include <math.h>

// ---------------- problem constants ----------------
#define NQ    80000      // queries (= 100*100*8 voxels)
#define CH    128        // channels
#define NH    8          // heads
#define HDIM  16         // head dim
#define NL    3          // levels
#define NPT   4          // points per level
#define SVAL  10080      // 48*160 + 24*80 + 12*40
#define FFD   512        // FFN hidden
#define NPTS_ 122880
#define SX 100
#define SY 100
#define SZ 8
#define PROJD 288        // 192 offsets + 96 attn logits
#define LN_EPS 1e-5f

// GEMM blocking
#define CT    32         // columns per block (2 col tiles, staged in LDS via TDM)
#define WAVES 4          // waves per block
#define MT    2          // row tiles per wave -> block covers 128 rows x 32 cols

typedef __attribute__((ext_vector_type(16))) __bf16 v16bf;
typedef __attribute__((ext_vector_type(8)))  __bf16 v8bf;
typedef __attribute__((ext_vector_type(8)))  float  v8f;
typedef __attribute__((ext_vector_type(4)))  unsigned int u32x4;
typedef __attribute__((ext_vector_type(4)))  int i32x4;
typedef __attribute__((ext_vector_type(8)))  int i32x8;

// ---------------- WMMA fragment loader ----------------
// 16-bit operand layout for V_WMMA_*_16X16X32 (ISA 7.12.2):
//   lanes 0-15 : row = lane,    VGPR0-3 hold K=0..7,  VGPR4-7 hold K=16..23
//   lanes 16-31: row = lane-16, VGPR0-3 hold K=8..15, VGPR4-7 hold K=24..31
// => per lane: two contiguous 8 x bf16 (16B) loads from a row-major matrix.
__device__ inline v16bf load_frag16(const __bf16* __restrict__ base, int ld,
                                    int row0, int k0, int lane) {
    int r  = lane & 15;
    int kh = lane >> 4;
    const __bf16* p = base + (size_t)(row0 + r) * ld + k0 + 8 * kh;
    union { v16bf v; v8bf h[2]; } u;
    u.h[0] = *(const v8bf*)(p);
    u.h[1] = *(const v8bf*)(p + 16);
    return u.v;
}

__device__ inline float gelu_exact(float x) {
    return 0.5f * x * (1.0f + erff(x * 0.70710678118654752f));
}

// ---------------- bf16 WMMA GEMM with TDM-staged B panel ----------------
// C(M x Nout) = A(M x K) * Bt(Nout x K)^T + bias
// EPI: 0 = f32 out, 1 = bf16 out, 2 = bf16 out with exact GELU
template <int EPI>
__global__ void k_gemm(const __bf16* __restrict__ A, const __bf16* __restrict__ Bt,
                       const float* __restrict__ bias,
                       float* __restrict__ Of, __bf16* __restrict__ Ob,
                       int M, int Nout, int K) {
    extern __shared__ __bf16 Bs[];               // CT x K weight panel
    const int lane   = threadIdx.x & 31;
    const int wv     = threadIdx.x >> 5;
    const int Mtiles = M >> 4;
    const int tn0    = blockIdx.x * (CT / 16);   // first col tile of this block
    const int tm0    = blockIdx.y * (WAVES * MT);

    // ---- stage the contiguous CT*K-element weight panel into LDS ----
#if __has_builtin(__builtin_amdgcn_tensor_load_to_lds)
    if (wv == 0) {
        // Tensor DMA descriptor (ISA 8.3/8.4): degenerate 1-D tile, 2-byte elems.
        const unsigned ctk   = (unsigned)(CT * K);            // tile elements (<= 16384)
        const size_t   gaddr = (size_t)(const void*)(Bt + (size_t)blockIdx.x * CT * K);
        const unsigned laddr = (unsigned)(size_t)(void*)Bs;   // LDS byte offset (low 32 bits)
        u32x4 g0;
        g0[0] = 1u;                                           // count = 1 (valid descriptor)
        g0[1] = laddr;                                        // lds_addr
        g0[2] = (unsigned)(gaddr & 0xFFFFFFFFu);              // global_addr[31:0]
        g0[3] = (unsigned)((gaddr >> 32) & 0x1FFFFFFu) | (2u << 30); // addr[56:32] | type=2
        const unsigned td0 = ctk;                             // tensor_dim0
        const unsigned td1 = 1u;                              // tensor_dim1
        i32x8 g1;
        g1[0] = (int)(1u << 16);                              // data_size = 1 (2 bytes)
        g1[1] = (int)((td0 & 0xFFFFu) << 16);                 // tensor_dim0[15:0]
        g1[2] = (int)(((td0 >> 16) & 0xFFFFu) | ((td1 & 0xFFFFu) << 16));
        g1[3] = (int)(((td1 >> 16) & 0xFFFFu) | (ctk << 16)); // tile_dim0 = CT*K
        g1[4] = (int)1;                                       // tile_dim1 = 1, tile_dim2 = 0
        g1[5] = (int)td0;                                     // tensor_dim0_stride lo32
        g1[6] = 0;
        g1[7] = 0;
        i32x4 z4 = {0, 0, 0, 0};
#if defined(__clang_major__) && (__clang_major__ >= 23)
        i32x8 z8 = {0, 0, 0, 0, 0, 0, 0, 0};
        __builtin_amdgcn_tensor_load_to_lds(g0, g1, z4, z4, z8, 0);
#else
        __builtin_amdgcn_tensor_load_to_lds(g0, g1, z4, z4, 0);
#endif
        __builtin_amdgcn_s_wait_tensorcnt(0);
    }
#else
    {
        const __bf16* src = Bt + (size_t)blockIdx.x * CT * K;
        int total = CT * K / 8;
        for (int i = threadIdx.x; i < total; i += blockDim.x)
            ((v8bf*)Bs)[i] = ((const v8bf*)src)[i];
    }
#endif
    __syncthreads();

    // row tiles of this wave (clamped; out-of-range tiles computed then discarded)
    const int tmw = tm0 + wv * MT;
    int tmr[MT];
#pragma unroll
    for (int i = 0; i < MT; ++i) tmr[i] = (tmw + i < Mtiles) ? (tmw + i) : 0;

    v8f acc[MT][2];
#pragma unroll
    for (int i = 0; i < MT; ++i)
#pragma unroll
        for (int t = 0; t < 2; ++t)
            acc[i][t] = (v8f){0.f, 0.f, 0.f, 0.f, 0.f, 0.f, 0.f, 0.f};

    for (int k0 = 0; k0 < K; k0 += 32) {
        v16bf b0 = load_frag16(Bs, K, 0,  k0, lane);   // ds_load_b128 pairs from LDS
        v16bf b1 = load_frag16(Bs, K, 16, k0, lane);
#pragma unroll
        for (int i = 0; i < MT; ++i) {
            v16bf a = load_frag16(A, K, tmr[i] * 16, k0, lane);
            acc[i][0] = __builtin_amdgcn_wmma_f32_16x16x32_bf16(
                false, a, false, b0, (short)0, acc[i][0], false, false);
            acc[i][1] = __builtin_amdgcn_wmma_f32_16x16x32_bf16(
                false, a, false, b1, (short)0, acc[i][1], false, false);
        }
    }

    // ---- epilogue: C/D layout VGPR r -> M = r + 8*(lane>>4), N = lane&15 ----
    int   ncol[2];
    float bv[2];
#pragma unroll
    for (int t = 0; t < 2; ++t) {
        ncol[t] = (tn0 + t) * 16 + (lane & 15);
        bv[t]   = bias ? bias[ncol[t]] : 0.f;
    }
#pragma unroll
    for (int i = 0; i < MT; ++i) {
        if (tmw + i >= Mtiles) continue;               // wave-uniform guard
        int mrow0 = (tmw + i) * 16 + 8 * (lane >> 4);
#pragma unroll
        for (int t = 0; t < 2; ++t)
#pragma unroll
            for (int r = 0; r < 8; ++r) {
                float v = acc[i][t][r] + bv[t];
                size_t idx = (size_t)(mrow0 + r) * Nout + ncol[t];
                if constexpr (EPI == 0) {
                    Of[idx] = v;
                } else if constexpr (EPI == 1) {
                    Ob[idx] = (__bf16)v;
                } else {
                    Ob[idx] = (__bf16)gelu_exact(v);
                }
            }
    }
}

// ---------------- prep kernels ----------------
__global__ void k_transpose_bf16(const float* __restrict__ W, __bf16* __restrict__ Wt,
                                 int indim, int outdim) {
    int idx = blockIdx.x * blockDim.x + threadIdx.x;
    if (idx >= indim * outdim) return;
    int o = idx / indim;
    int i = idx - o * indim;
    Wt[idx] = (__bf16)W[(size_t)i * outdim + o];   // Wt[o][i] = W[i][o]
}

__global__ void k_prep_q(const float* __restrict__ se, const float* __restrict__ pos,
                         __bf16* __restrict__ qb) {
    size_t idx = (size_t)blockIdx.x * blockDim.x + threadIdx.x;
    if (idx >= (size_t)NQ * CH) return;
    qb[idx] = (__bf16)(se[idx] + pos[idx]);
}

__global__ void k_build_vals(const float* __restrict__ f0, const float* __restrict__ f1,
                             const float* __restrict__ f2, __bf16* __restrict__ valsb) {
    int idx = blockIdx.x * blockDim.x + threadIdx.x;
    if (idx >= SVAL * CH) return;
    int s = idx / CH;
    int c = idx - s * CH;
    float v;
    if (s < 7680)       v = f0[c * 7680 + s];
    else if (s < 9600)  v = f1[c * 1920 + (s - 7680)];
    else                v = f2[c * 480  + (s - 9600)];
    valsb[idx] = (__bf16)v;
}

__global__ void k_concat_bias(const float* __restrict__ b0, const float* __restrict__ b1,
                              float* __restrict__ dst) {
    int i = blockIdx.x * blockDim.x + threadIdx.x;
    if (i < 192)      dst[i] = b0[i];
    else if (i < 288) dst[i] = b1[i - 192];
}

__global__ void k_mask_clear(unsigned* __restrict__ mask) {
    int i = blockIdx.x * blockDim.x + threadIdx.x;
    if (i < NQ) mask[i] = 0u;
}

__global__ void k_mask_set(const int* __restrict__ vol, unsigned* __restrict__ mask) {
    int i = blockIdx.x * blockDim.x + threadIdx.x;
    if (i >= NPTS_) return;
    int x = vol[3 * i], y = vol[3 * i + 1], z = vol[3 * i + 2];
    if (x >= 0 && x < SX && y >= 0 && y < SY && z >= 0 && z < SZ) {
        int flat = (x * SY + y) * SZ + z;
        atomicOr(&mask[flat], 1u);
    }
}

// ---------------- deformable sampling: one thread = (query, head) ----------------
__global__ void k_sample(const float* __restrict__ proj, const float* __restrict__ ref,
                         const __bf16* __restrict__ vb, __bf16* __restrict__ sampb) {
    int idx = blockIdx.x * blockDim.x + threadIdx.x;
    if (idx >= NQ * NH) return;
    int n = idx >> 3;
    int h = idx & 7;
    const float* pr = proj + (size_t)n * PROJD;

    // softmax over 12 attn logits for this head
    float lg[NL * NPT];
    float mx = -3.4e38f;
#pragma unroll
    for (int j = 0; j < NL * NPT; ++j) {
        lg[j] = pr[192 + h * (NL * NPT) + j];
        mx = fmaxf(mx, lg[j]);
    }
    float den = 0.f;
#pragma unroll
    for (int j = 0; j < NL * NPT; ++j) { lg[j] = __expf(lg[j] - mx); den += lg[j]; }
    float inv = 1.f / den;

    float rx = ref[2 * n], ry = ref[2 * n + 1];
    float acc[HDIM];
#pragma unroll
    for (int d = 0; d < HDIM; ++d) acc[d] = 0.f;

    const int lw[NL]  = {160, 80, 40};
    const int lh[NL]  = {48, 24, 12};
    const int lst[NL] = {0, 7680, 9600};

    for (int l = 0; l < NL; ++l) {
        int Wl = lw[l], Hl = lh[l], st = lst[l];
        float fW = (float)Wl, fH = (float)Hl;
        for (int p = 0; p < NPT; ++p) {
            float ox = pr[((h * NL + l) * NPT + p) * 2 + 0];
            float oy = pr[((h * NL + l) * NPT + p) * 2 + 1];
            // x = (rx + ox/W)*W - 0.5 = rx*W + ox - 0.5
            float x = rx * fW + ox - 0.5f;
            float y = ry * fH + oy - 0.5f;
            float x0 = floorf(x), y0 = floorf(y);
            float aw = lg[l * NPT + p] * inv;
#pragma unroll
            for (int dx = 0; dx < 2; ++dx)
#pragma unroll
                for (int dy = 0; dy < 2; ++dy) {
                    float xi = x0 + dx, yi = y0 + dy;
                    if (xi < 0.f || xi >= fW || yi < 0.f || yi >= fH) continue;
                    float wgt = (1.f - fabsf(x - xi)) * (1.f - fabsf(y - yi)) * aw;
                    int xc = (int)xi, yc = (int)yi;
                    const v8bf* gp = (const v8bf*)(vb + (size_t)(st + yc * Wl + xc) * CH + h * HDIM);
                    v8bf g0 = gp[0];
                    v8bf g1 = gp[1];
#pragma unroll
                    for (int d = 0; d < 8; ++d) {
                        acc[d]     += wgt * (float)g0[d];
                        acc[8 + d] += wgt * (float)g1[d];
                    }
                }
        }
    }
    __bf16* op = sampb + (size_t)n * CH + h * HDIM;
#pragma unroll
    for (int d = 0; d < HDIM; ++d) op[d] = (__bf16)acc[d];
}

// ---------------- wave-per-row LayerNorm (wave32, 4 channels / lane) ----------------
__global__ void k_ln1(const float* __restrict__ se, const float* __restrict__ att,
                      const float* __restrict__ g, const float* __restrict__ b,
                      float* __restrict__ xqf, __bf16* __restrict__ xqb) {
    int wid  = blockIdx.x * (blockDim.x >> 5) + (threadIdx.x >> 5);
    int lane = threadIdx.x & 31;
    if (wid >= NQ) return;
    size_t base = (size_t)wid * CH;
    float v[4];
    float s = 0.f;
#pragma unroll
    for (int i = 0; i < 4; ++i) {
        int c = lane + 32 * i;
        v[i] = se[base + c] + att[base + c];
        s += v[i];
    }
#pragma unroll
    for (int m = 16; m >= 1; m >>= 1) s += __shfl_xor(s, m, 32);
    float mean = s * (1.0f / CH);
    float vs = 0.f;
#pragma unroll
    for (int i = 0; i < 4; ++i) { float d = v[i] - mean; vs += d * d; }
#pragma unroll
    for (int m = 16; m >= 1; m >>= 1) vs += __shfl_xor(vs, m, 32);
    float rinv = rsqrtf(vs * (1.0f / CH) + LN_EPS);
#pragma unroll
    for (int i = 0; i < 4; ++i) {
        int c = lane + 32 * i;
        float r = (v[i] - mean) * rinv * g[c] + b[c];
        xqf[base + c] = r;
        xqb[base + c] = (__bf16)r;
    }
}

// LN2 + mask select + transposed scatter to (C, X, Y, Z)
__global__ void k_final(const float* __restrict__ xqf, const float* __restrict__ ffn,
                        const float* __restrict__ se,
                        const float* __restrict__ g, const float* __restrict__ b,
                        const unsigned* __restrict__ mask, float* __restrict__ out) {
    int wid  = blockIdx.x * (blockDim.x >> 5) + (threadIdx.x >> 5);
    int lane = threadIdx.x & 31;
    if (wid >= NQ) return;
    size_t base = (size_t)wid * CH;
    float v[4];
    float s = 0.f;
#pragma unroll
    for (int i = 0; i < 4; ++i) {
        int c = lane + 32 * i;
        v[i] = xqf[base + c] + ffn[base + c];
        s += v[i];
    }
#pragma unroll
    for (int m = 16; m >= 1; m >>= 1) s += __shfl_xor(s, m, 32);
    float mean = s * (1.0f / CH);
    float vs = 0.f;
#pragma unroll
    for (int i = 0; i < 4; ++i) { float d = v[i] - mean; vs += d * d; }
#pragma unroll
    for (int m = 16; m >= 1; m >>= 1) vs += __shfl_xor(vs, m, 32);
    float rinv = rsqrtf(vs * (1.0f / CH) + LN_EPS);
    bool keep = mask[wid] != 0u;
#pragma unroll
    for (int i = 0; i < 4; ++i) {
        int c = lane + 32 * i;
        float r = (v[i] - mean) * rinv * g[c] + b[c];
        if (!keep) r = se[base + c];
        out[(size_t)c * NQ + wid] = r;   // (C, X*Y*Z) layout
    }
}

// ---------------- host launcher ----------------
extern "C" void kernel_launch(void* const* d_in, const int* in_sizes, int n_in,
                              void* d_out, int out_size, void* d_ws, size_t ws_size,
                              hipStream_t stream) {
    const float* se    = (const float*)d_in[0];
    const float* pos   = (const float*)d_in[1];
    const float* feat0 = (const float*)d_in[2];
    const float* feat1 = (const float*)d_in[3];
    const float* feat2 = (const float*)d_in[4];
    const int*   vol   = (const int*)d_in[5];
    const float* ref   = (const float*)d_in[6];
    const float* Woff  = (const float*)d_in[7];
    const float* boff  = (const float*)d_in[8];
    const float* Wattn = (const float*)d_in[9];
    const float* battn = (const float*)d_in[10];
    const float* Wval  = (const float*)d_in[11];
    const float* bval  = (const float*)d_in[12];
    const float* Wout  = (const float*)d_in[13];
    const float* bout  = (const float*)d_in[14];
    const float* ln1g  = (const float*)d_in[15];
    const float* ln1b  = (const float*)d_in[16];
    const float* ln2g  = (const float*)d_in[17];
    const float* ln2b  = (const float*)d_in[18];
    const float* Wff1  = (const float*)d_in[19];
    const float* bff1  = (const float*)d_in[20];
    const float* Wff2  = (const float*)d_in[21];
    const float* bff2  = (const float*)d_in[22];
    float* out = (float*)d_out;

    // -------- workspace carve-out (256B aligned) --------
    char* ws = (char*)d_ws;
    size_t o = 0;
    auto carve = [&](size_t bytes) {
        size_t cur = o;
        o += (bytes + 255) & ~(size_t)255;
        return (void*)(ws + cur);
    };
    __bf16* qb     = (__bf16*)carve((size_t)NQ * CH * 2);        // q = embed+pos, bf16
    __bf16* valsb  = (__bf16*)carve((size_t)SVAL * CH * 2);      // feats as (S, C) bf16
    __bf16* vbuf   = (__bf16*)carve((size_t)SVAL * CH * 2);      // v = vals@Wv + b
    float*  proj   = (float*)carve((size_t)NQ * PROJD * 4);      // off|attn (also reused f32 buf)
    __bf16* sampb  = (__bf16*)carve((size_t)NQ * CH * 2);        // sampled attn output
    float*  xqf    = (float*)carve((size_t)NQ * CH * 4);         // LN1 result f32
    __bf16* xqb    = (__bf16*)carve((size_t)NQ * CH * 2);        // LN1 result bf16
    __bf16* h1b    = (__bf16*)carve((size_t)NQ * FFD * 2);       // GELU(FFN hidden) bf16
    unsigned* mask = (unsigned*)carve((size_t)NQ * 4);
    __bf16* Wvt    = (__bf16*)carve((size_t)CH * CH * 2);        // transposed bf16 weights
    __bf16* Wcatt  = (__bf16*)carve((size_t)PROJD * CH * 2);
    __bf16* Woutt  = (__bf16*)carve((size_t)CH * CH * 2);
    __bf16* Wff1t  = (__bf16*)carve((size_t)FFD * CH * 2);
    __bf16* Wff2t  = (__bf16*)carve((size_t)CH * FFD * 2);
    float*  bcat   = (float*)carve((size_t)PROJD * 4);
    float*  attnf  = proj;   // reuse: attn_out f32 (N x 128)
    float*  ffn2f  = proj;   // reuse: ffn2 out f32 (N x 128)
    (void)ws_size; (void)in_sizes; (void)n_in; (void)out_size;

    const int TB = 256;
    auto blocks = [](size_t n, int tb) { return (int)((n + tb - 1) / tb); };

    // -------- weight prep (bf16 transposed: Nout x K) --------
    k_transpose_bf16<<<blocks(CH * CH, TB), TB, 0, stream>>>(Wval, Wvt, CH, CH);
    k_transpose_bf16<<<blocks(CH * 192, TB), TB, 0, stream>>>(Woff, Wcatt, CH, 192);
    k_transpose_bf16<<<blocks(CH * 96, TB), TB, 0, stream>>>(Wattn, Wcatt + 192 * CH, CH, 96);
    k_transpose_bf16<<<blocks(CH * CH, TB), TB, 0, stream>>>(Wout, Woutt, CH, CH);
    k_transpose_bf16<<<blocks(CH * FFD, TB), TB, 0, stream>>>(Wff1, Wff1t, CH, FFD);
    k_transpose_bf16<<<blocks(FFD * CH, TB), TB, 0, stream>>>(Wff2, Wff2t, FFD, CH);
    k_concat_bias<<<blocks(PROJD, TB), TB, 0, stream>>>(boff, battn, bcat);

    // -------- activation prep --------
    k_prep_q<<<blocks((size_t)NQ * CH, TB), TB, 0, stream>>>(se, pos, qb);
    k_build_vals<<<blocks((size_t)SVAL * CH, TB), TB, 0, stream>>>(feat0, feat1, feat2, valsb);
    k_mask_clear<<<blocks(NQ, TB), TB, 0, stream>>>(mask);
    k_mask_set<<<blocks(NPTS_, TB), TB, 0, stream>>>(vol, mask);

    // GEMM launcher: block = 128 threads (4 waves), covers 128 rows x CT cols,
    // LDS = CT*K bf16 weight panel staged by TDM.
#define LAUNCH_GEMM(EPI, Aa, Bb, bb, Ofp, Obp, Mm, Nn, Kk)                           \
    do {                                                                             \
        int mtiles  = (Mm) / 16;                                                     \
        int gy      = (mtiles + WAVES * MT - 1) / (WAVES * MT);                      \
        dim3 gg((Nn) / CT, gy);                                                      \
        size_t shb  = (size_t)CT * (Kk) * 2;                                         \
        k_gemm<EPI><<<gg, WAVES * 32, shb, stream>>>(Aa, Bb, bb, Ofp, Obp, Mm, Nn, Kk); \
    } while (0)

    // v = vals @ W_val + b_val  (bf16 out)
    LAUNCH_GEMM(1, valsb, Wvt, bval, nullptr, vbuf, SVAL, CH, CH);

    // proj = q @ [W_off | W_attn] + [b_off | b_attn]  (f32 out)
    LAUNCH_GEMM(0, qb, Wcatt, bcat, proj, nullptr, NQ, PROJD, CH);

    // deformable sampling (softmax + bilinear gather)
    k_sample<<<blocks((size_t)NQ * NH, TB), TB, 0, stream>>>(proj, ref, vbuf, sampb);

    // attn_out = samp @ W_out + b_out  (f32, reuses proj buffer)
    LAUNCH_GEMM(0, sampb, Woutt, bout, attnf, nullptr, NQ, CH, CH);

    // xq = LN1(embed + attn_out)
    k_ln1<<<blocks((size_t)NQ * 32, TB), TB, 0, stream>>>(se, attnf, ln1g, ln1b, xqf, xqb);

    // h1 = gelu(xq @ W_ff1 + b_ff1)  (bf16 out)
    LAUNCH_GEMM(2, xqb, Wff1t, bff1, nullptr, h1b, NQ, FFD, CH);

    // ffn2 = h1 @ W_ff2 + b_ff2  (f32, reuses proj buffer)
    LAUNCH_GEMM(0, h1b, Wff2t, bff2, ffn2f, nullptr, NQ, CH, FFD);

    // LN2 + mask select + transpose to (C, X, Y, Z)
    k_final<<<blocks((size_t)NQ * 32, TB), TB, 0, stream>>>(
        xqf, ffn2f, se, ln2g, ln2b, mask, out);
#undef LAUNCH_GEMM
}
